// channel_crossmodel_91096256348671
// MI455X (gfx1250) — matrix operand order repaired and checked by reading, not compiled, not run on hardware
//
#include <hip/hip_runtime.h>
#include <math.h>

// Problem constants (from reference setup_inputs)
#define BB 4
#define CC 512        // FFT length (channel axis)
#define LL 2048
#define KN 4
#define KS 7
#define NT 256        // threads per block = 8 wave32 waves

typedef __attribute__((ext_vector_type(16))) _Float16 v16h;
typedef __attribute__((ext_vector_type(8)))  float    v8f;

// ---------- complex helpers ----------
__device__ __forceinline__ float2 cadd(float2 a, float2 b) { return make_float2(a.x + b.x, a.y + b.y); }
__device__ __forceinline__ float2 csub(float2 a, float2 b) { return make_float2(a.x - b.x, a.y - b.y); }
__device__ __forceinline__ float2 cmul(float2 a, float2 b) {
  return make_float2(a.x * b.x - a.y * b.y, a.x * b.y + a.y * b.x);
}
__device__ __forceinline__ float2 csqrtf2(float2 z) {
  float r  = sqrtf(z.x * z.x + z.y * z.y);
  float re = sqrtf(fmaxf(0.5f * (r + z.x), 0.0f));
  float im = copysignf(sqrtf(fmaxf(0.5f * (r - z.x), 0.0f)), z.y);
  return make_float2(re, im);
}

// ---------- 512-point radix-2 DIT FFT in LDS (input pre-bit-reversed) ----------
__device__ __forceinline__ void fft512(float2* X, float sgn, int tid) {
  #pragma unroll
  for (int s = 1; s <= 9; ++s) {
    int half = 1 << (s - 1);
    int grp  = tid >> (s - 1);
    int pos  = tid & (half - 1);
    int i0   = (grp << s) + pos;
    int i1   = i0 + half;
    float ang = sgn * 6.28318530717958647f * (float)pos / (float)(1 << s);
    float sw, cw;
    __sincosf(ang, &sw, &cw);
    float2 u = X[i0];
    float2 t = X[i1];
    float2 wt = make_float2(t.x * cw - t.y * sw, t.x * sw + t.y * cw);
    X[i0] = make_float2(u.x + wt.x, u.y + wt.y);
    X[i1] = make_float2(u.x - wt.x, u.y - wt.y);
    __syncthreads();
  }
}

// complex block-wide sum over the 256 per-thread values; all threads get result
__device__ __forceinline__ float2 block_reduce_c(float2 v, float2* red, int tid) {
  red[tid] = v;
  __syncthreads();
  #pragma unroll
  for (int off = 128; off > 0; off >>= 1) {
    if (tid < off) {
      red[tid].x += red[tid + off].x;
      red[tid].y += red[tid + off].y;
    }
    __syncthreads();
  }
  float2 r = red[0];
  __syncthreads();
  return r;
}

// ---------- qmean[b,l] = mean over c of q ----------
__global__ __launch_bounds__(NT) void qmean_kernel(const float* __restrict__ q,
                                                   float* __restrict__ qmean) {
  int idx = blockIdx.x * NT + threadIdx.x;     // 0 .. B*L-1 (8192)
  int b = idx >> 11;
  int l = idx & (LL - 1);
  const float* qp = q + (size_t)b * CC * LL + l;
  float s = 0.0f;
  for (int c = 0; c < CC; ++c) s += qp[(size_t)c * LL];
  qmean[idx] = s * (1.0f / (float)CC);
}

// ---------- fused: conv7 -> FFT_C -> complex filter -> IFFT_C -> |.| ----------
// one block per (b,l) column; outs stored as (B,K,L,C) so c-stores coalesce
__global__ __launch_bounds__(NT) void main_kernel(const float* __restrict__ x,
                                                  const float* __restrict__ q,
                                                  const float* __restrict__ sigmas,
                                                  const float* __restrict__ centers,
                                                  float* __restrict__ outs) {
  __shared__ float  kern[KN][KS];
  __shared__ float2 qf[CC];
  __shared__ float2 work[CC];
  __shared__ float2 tmp[CC];
  __shared__ float2 red[NT];

  int tid = threadIdx.x;
  int b   = blockIdx.x >> 11;          // / LL
  int l   = blockIdx.x & (LL - 1);

  // Gaussian kernels (4 threads, one per k)
  if (tid < KN) {
    float sg = sigmas[tid] + 0.001f;
    float ce = centers[tid];
    float pdf[KS];
    float s = 0.0f;
    #pragma unroll
    for (int t = 0; t < KS; ++t) {
      float z = ((float)t - 3.0f - ce) / sg;
      pdf[t] = __expf(-0.5f * z * z);
      s += pdf[t];
    }
    float inv = 1.0f / (s + 0.01f);
    #pragma unroll
    for (int t = 0; t < KS; ++t) kern[tid][t] = pdf[t] * inv;
  }

  // load q column (bit-reversed) and forward FFT
  for (int e = tid; e < CC; e += NT) {
    int c = (int)(__brev((unsigned)e) >> 23);
    work[e] = make_float2(q[((size_t)b * CC + c) * LL + l], 0.0f);
  }
  __syncthreads();
  fft512(work, -1.0f, tid);

  float2 q0 = work[tid], q1 = work[tid + NT];
  qf[tid] = q0; qf[tid + NT] = q1;
  float2 Sq = block_reduce_c(cadd(q0, q1), red, tid);
  float2 my = make_float2(Sq.x * (1.0f / CC), Sq.y * (1.0f / CC));
  float2 dy0 = csub(q0, my), dy1 = csub(q1, my);
  float2 bvar = block_reduce_c(cadd(cmul(dy0, dy0), cmul(dy1, dy1)), red, tid);

  for (int k = 0; k < KN; ++k) {
    __syncthreads();
    // depthwise conv (edge-clamped) loaded bit-reversed for FFT
    for (int e = tid; e < CC; e += NT) {
      int c = (int)(__brev((unsigned)e) >> 23);
      const float* xr = x + ((size_t)b * CC + c) * LL;
      __builtin_prefetch(xr + l, 0, 3);   // global_prefetch_b8 (L2-resident x)
      float acc = 0.0f;
      #pragma unroll
      for (int t = 0; t < KS; ++t) {
        int li = l + t - 3;
        li = min(max(li, 0), LL - 1);
        acc += kern[k][t] * xr[li];
      }
      work[e] = make_float2(acc, 0.0f);
    }
    __syncthreads();
    fft512(work, -1.0f, tid);

    float2 v0 = work[tid], v1 = work[tid + NT];
    float2 Sv = block_reduce_c(cadd(v0, v1), red, tid);
    float2 mx = make_float2(Sv.x * (1.0f / CC), Sv.y * (1.0f / CC));
    float2 e0 = csub(v0, mx), e1 = csub(v1, mx);
    float2 avar = block_reduce_c(cadd(cmul(e0, e0), cmul(e1, e1)), red, tid);

    float2 sab = cmul(avar, bvar);
    sab.x += 0.0001f;
    float2 sq = csqrtf2(sab);
    float den = sq.x * sq.x + sq.y * sq.y;
    float dinv = 1.0f / den;

    // elementwise frequency-domain blend
    for (int c2 = tid; c2 < CC; c2 += NT) {
      float2 vf = work[c2];
      float2 qc = qf[c2];
      float2 dx = csub(vf, mx);
      float2 dyc = csub(qc, my);
      float2 cov = cmul(dx, dyc);
      float2 rho = make_float2((cov.x * sq.x + cov.y * sq.y) * dinv,
                               (cov.y * sq.x - cov.x * sq.y) * dinv);
      float vimag = atan2f(rho.y, rho.x + 0.0001f);
      float wm = tanhf(0.25f * vimag);   // == 2/(1+exp(-0.5 v)) - 1
      if (!(wm == wm)) wm = 0.0f;        // NaN -> 0
      float fx = vf.x - qc.x, fy = vf.y - qc.y;
      float w = __expf(-0.5f * (fx * fx + fy * fy)) * wm;
      tmp[c2] = make_float2(qc.x * (1.0f - w) + vf.x * w,
                            qc.y * (1.0f - w) + vf.y * w);
    }
    __syncthreads();
    // bit-reverse permute for inverse FFT
    for (int e = tid; e < CC; e += NT) {
      int c2 = (int)(__brev((unsigned)e) >> 23);
      work[e] = tmp[c2];
    }
    __syncthreads();
    fft512(work, +1.0f, tid);

    // |ifft| * 1/N  -> outs (B,K,L,C): coalesced over c
    float* op = outs + ((size_t)(b * KN + k) * LL + l) * CC;
    for (int c2 = tid; c2 < CC; c2 += NT) {
      float2 r = work[c2];
      op[c2] = sqrtf(r.x * r.x + r.y * r.y) * (1.0f / (float)CC);
    }
  }
}

// ---------- scores[b,c,k] = sum_l qmean[b,l] * outs[b,k,l,c]  (WMMA f16->f32) ----------
// one wave per (b,k,c-tile16); B replicated across N so every output column
// equals the matvec result; layouts per CDNA5 ISA §7.12.2.
__global__ __launch_bounds__(NT) void scores_kernel(const float* __restrict__ outs,
                                                    const float* __restrict__ qmean,
                                                    float* __restrict__ scores) {
  int tid  = threadIdx.x;
  int lane = tid & 31;
  int wave = (blockIdx.x << 3) + (tid >> 5);   // 0..511
  int ctile = wave & 31;
  int k     = (wave >> 5) & 3;
  int b     = wave >> 7;
  int c0 = ctile << 4;
  int m  = lane & 15;
  int hi = lane >> 4;

  const float* base = outs + (size_t)(b * KN + k) * LL * CC;
  const float* qm   = qmean + b * LL;

  v8f acc = {};
  for (int l0 = 0; l0 < LL; l0 += 32) {
    v16h a, bv;
    #pragma unroll
    for (int j = 0; j < 8; ++j) {
      int K0 = 8 * hi + j;            // A 16x32 f16: lanes0-15 h[0..7]=K0..7
      int K1 = 16 + 8 * hi + j;       //              h[8..15]=K16..23 (hi mirrors)
      a[j]     = (_Float16)base[(size_t)(l0 + K0) * CC + c0 + m];
      a[j + 8] = (_Float16)base[(size_t)(l0 + K1) * CC + c0 + m];
    }
    #pragma unroll
    for (int j = 0; j < 16; ++j) {    // B 32x16 f16: j<8 -> K=8*hi+j, j>=8 -> K=16+8*hi+(j-8)
      bv[j] = (_Float16)qm[l0 + (j < 8 ? 8 * hi + j : 16 + 8 * hi + (j - 8))];
    }
    acc = __builtin_amdgcn_wmma_f32_16x16x32_f16(
        /*neg_a=*/false, a, /*neg_b=*/false, bv,
        /*c_mod=*/(short)0, acc, /*reuse_a=*/false, /*reuse_b=*/false);
  }
  // D 16x16 f32 layout: lane0 holds M=0..7 (N=0) in acc[0..7]; lane16 holds M=8..15
  if (lane == 0 || lane == 16) {
    int rbase = (lane == 0) ? 0 : 8;
    #pragma unroll
    for (int v = 0; v < 8; ++v) {
      scores[((size_t)b * CC + c0 + rbase + v) * KN + k] = acc[v];
    }
  }
}

// ---------- softmax over k + weighted sum -> out (B,C,L) ----------
__global__ __launch_bounds__(NT) void final_kernel(const float* __restrict__ outs,
                                                   const float* __restrict__ scores,
                                                   float* __restrict__ out) {
  int bc = blockIdx.x;                 // 0 .. B*C-1
  int b = bc >> 9;
  int c = bc & (CC - 1);

  float s0 = scores[(size_t)bc * KN + 0];
  float s1 = scores[(size_t)bc * KN + 1];
  float s2 = scores[(size_t)bc * KN + 2];
  float s3 = scores[(size_t)bc * KN + 3];
  float mx = fmaxf(fmaxf(s0, s1), fmaxf(s2, s3));
  float e0 = __expf(s0 - mx), e1 = __expf(s1 - mx);
  float e2 = __expf(s2 - mx), e3 = __expf(s3 - mx);
  float inv = 1.0f / (e0 + e1 + e2 + e3);
  float w0 = e0 * inv, w1 = e1 * inv, w2 = e2 * inv, w3 = e3 * inv;

  const float* o0 = outs + (size_t)(b * KN + 0) * LL * CC + c;
  const float* o1 = outs + (size_t)(b * KN + 1) * LL * CC + c;
  const float* o2 = outs + (size_t)(b * KN + 2) * LL * CC + c;
  const float* o3 = outs + (size_t)(b * KN + 3) * LL * CC + c;

  for (int l = threadIdx.x; l < LL; l += NT) {
    size_t off = (size_t)l * CC;       // L2-resident gathers
    out[(size_t)bc * LL + l] =
        w0 * o0[off] + w1 * o1[off] + w2 * o2[off] + w3 * o3[off];
  }
}

extern "C" void kernel_launch(void* const* d_in, const int* in_sizes, int n_in,
                              void* d_out, int out_size, void* d_ws, size_t ws_size,
                              hipStream_t stream) {
  (void)in_sizes; (void)n_in; (void)out_size; (void)ws_size;
  const float* x       = (const float*)d_in[0];
  const float* q       = (const float*)d_in[1];
  const float* sigmas  = (const float*)d_in[2];
  const float* centers = (const float*)d_in[3];
  // d_in[4] = kernel_size (int scalar) == 7, compile-time constant here

  // workspace layout (floats): outs (B*K*L*C) | qmean (B*L) | scores (B*C*K)
  float* ws     = (float*)d_ws;
  float* outs   = ws;                                    // 16,777,216 floats
  float* qmean  = ws + (size_t)BB * KN * LL * CC;        // 8,192 floats
  float* scores = qmean + (size_t)BB * LL;               // 8,192 floats
  float* out    = (float*)d_out;

  qmean_kernel<<<(BB * LL) / NT, NT, 0, stream>>>(q, qmean);
  main_kernel<<<BB * LL, NT, 0, stream>>>(x, q, sigmas, centers, outs);
  scores_kernel<<<(BB * KN * (CC / 16)) / 8, NT, 0, stream>>>(outs, qmean, scores);
  final_kernel<<<BB * CC, NT, 0, stream>>>(outs, scores, out);
}
